// CausalAttention_27032524161192
// MI455X (gfx1250) — matrix operand order repaired and checked by reading, not compiled, not run on hardware
//
#include <hip/hip_runtime.h>

#define DEVFN __device__ __forceinline__

typedef __attribute__((ext_vector_type(16))) __bf16       v16bf;
typedef __attribute__((ext_vector_type(8)))  float        v8f;
typedef __attribute__((ext_vector_type(4)))  unsigned int u32x4;
typedef __attribute__((ext_vector_type(4)))  int          i32x4;
typedef __attribute__((ext_vector_type(4)))  float        f32x4;

constexpr int CB = 2, CL = 2048, CD = 2048, CH = 32, CHD = 64;

// ---------------- helpers ----------------

DEVFN __bf16 f2bf(float f) {
    union { float f; unsigned int u; } in; in.f = f;
    unsigned int u = in.u;
    unsigned int r = u + 0x7FFFu + ((u >> 16) & 1u);  // round-to-nearest-even
    union { unsigned short s; __bf16 b; } out;
    out.s = (unsigned short)(r >> 16);
    return out.b;
}

union FragBF { v16bf v; u32x4 q[2]; };

// A-fragment (16-bit A 16x32 layout): elems 0..7 = K[k0..k0+7], 8..15 = K[k0+16..k0+23],
// where caller passes k0 = kbase + 8*half.
DEVFN v16bf frag_a(const __bf16* row, int k0) {
    FragBF f;
    f.q[0] = *(const u32x4*)(row + k0);
    f.q[1] = *(const u32x4*)(row + k0 + 16);
    return f.v;
}
// B-fragment (B 32x16 layout, lane=column): elems 0..7 = K[k0..k0+7], 8..15 = K[k0+16..k0+23],
// caller passes k0 = kbase + 8*half.
DEVFN v16bf frag_b(const __bf16* row, int k0) {
    FragBF f;
    f.q[0] = *(const u32x4*)(row + k0);
    f.q[1] = *(const u32x4*)(row + k0 + 16);
    return f.v;
}

DEVFN v8f wmma_bf16(v16bf a, v16bf b, v8f c) {
    return __builtin_amdgcn_wmma_f32_16x16x32_bf16(
        /*neg_a=*/false, a, /*neg_b=*/false, b,
        /*c_mod=*/(short)0, c, /*reuse_a=*/false, /*reuse_b=*/false);
}

DEVFN float halfmax16(float v) {   // reduce over 16-lane half (rows live per half)
#pragma unroll
    for (int m = 1; m <= 8; m <<= 1) v = fmaxf(v, __shfl_xor(v, m, 32));
    return v;
}
DEVFN float halfsum16(float v) {
#pragma unroll
    for (int m = 1; m <= 8; m <<= 1) v += __shfl_xor(v, m, 32);
    return v;
}

// ---------------- optional async global->LDS path ----------------
#if defined(__has_builtin)
#if __has_builtin(__builtin_amdgcn_global_load_async_to_lds_b128) && \
    __has_builtin(__builtin_amdgcn_s_wait_asynccnt)
#define ATTN_ASYNC_COPY 1
#endif
#endif

#ifdef ATTN_ASYNC_COPY
typedef __attribute__((address_space(1))) i32x4 gas_i32x4;  // global
typedef __attribute__((address_space(3))) i32x4 las_i32x4;  // LDS
#endif

DEVFN void cp16_g2l(__bf16* dst_lds, const __bf16* src_g) {
#ifdef ATTN_ASYNC_COPY
    __builtin_amdgcn_global_load_async_to_lds_b128(
        (gas_i32x4*)src_g, (las_i32x4*)dst_lds, 0, 0);
#else
    *(u32x4*)dst_lds = *(const u32x4*)src_g;
#endif
}
DEVFN void cp_wait() {
#ifdef ATTN_ASYNC_COPY
    __builtin_amdgcn_s_wait_asynccnt(0);
#endif
}

// ---------------- kernel: fp32 -> bf16 convert ----------------

__global__ __launch_bounds__(256) void cvt_f32_bf16(
    const float* __restrict__ in, __bf16* __restrict__ out, int n4) {
    int i = blockIdx.x * 256 + threadIdx.x;
    if (i >= n4) return;
    f32x4 v = ((const f32x4*)in)[i];
    union { __bf16 b[4]; unsigned long long u; } p;
    p.b[0] = f2bf(v.x); p.b[1] = f2bf(v.y); p.b[2] = f2bf(v.z); p.b[3] = f2bf(v.w);
    ((unsigned long long*)out)[i] = p.u;
}

// ---------------- kernel: GEMM  out = A(M x 2048) @ W(2048 x 2048)^T ----------------
// Block: 256 threads = 8 waves arranged 4(M) x 2(N); block tile 128x128.
// Wave tile: 32(M) x 64(N) = 2x4 WMMA tiles -> 8 WMMA per 6 fragment loads per k-step.
// MODE 0: out = Qh/Kh layout (B,H,L,HD) bf16
// MODE 1: out = Vt layout (B,H,HD,L) bf16  (transposed V projection)
// MODE 2: out = fp32 row-major (M x 2048)
template <int MODE>
__global__ __launch_bounds__(256) void gemm_bf16_nt(
    const __bf16* __restrict__ A, const __bf16* __restrict__ W,
    void* __restrict__ outv) {
    const int tid  = threadIdx.x;
    const int wave = __builtin_amdgcn_readfirstlane(tid >> 5);  // 0..7
    const int lane = tid & 31;
    const int r    = lane & 15;
    const int hf   = lane >> 4;
    const int mw   = blockIdx.y * 128 + (wave >> 1) * 32;  // 4 wave-rows
    const int nw   = blockIdx.x * 128 + (wave & 1) * 64;   // 2 wave-cols

    const __bf16* a0 = A + (size_t)(mw + r) * CD;
    const __bf16* a1 = A + (size_t)(mw + 16 + r) * CD;
    const __bf16* bw0 = W + (size_t)(nw + r) * CD;
    const __bf16* bw1 = W + (size_t)(nw + 16 + r) * CD;
    const __bf16* bw2 = W + (size_t)(nw + 32 + r) * CD;
    const __bf16* bw3 = W + (size_t)(nw + 48 + r) * CD;
    const int ka = 8 * hf, kb = 8 * hf;

    v8f zf = {0.f,0.f,0.f,0.f,0.f,0.f,0.f,0.f};
    v8f acc[2][4];
#pragma unroll
    for (int i = 0; i < 2; ++i)
#pragma unroll
        for (int j = 0; j < 4; ++j) acc[i][j] = zf;

    for (int kk = 0; kk < CD; kk += 32) {
        v16bf af0 = frag_a(a0, kk + ka);
        v16bf af1 = frag_a(a1, kk + ka);
        v16bf bf0 = frag_b(bw0, kk + kb);
        v16bf bf1 = frag_b(bw1, kk + kb);
        v16bf bf2 = frag_b(bw2, kk + kb);
        v16bf bf3 = frag_b(bw3, kk + kb);
        acc[0][0] = wmma_bf16(af0, bf0, acc[0][0]);
        acc[0][1] = wmma_bf16(af0, bf1, acc[0][1]);
        acc[0][2] = wmma_bf16(af0, bf2, acc[0][2]);
        acc[0][3] = wmma_bf16(af0, bf3, acc[0][3]);
        acc[1][0] = wmma_bf16(af1, bf0, acc[1][0]);
        acc[1][1] = wmma_bf16(af1, bf1, acc[1][1]);
        acc[1][2] = wmma_bf16(af1, bf2, acc[1][2]);
        acc[1][3] = wmma_bf16(af1, bf3, acc[1][3]);
    }

#pragma unroll
    for (int i = 0; i < 2; ++i)
#pragma unroll
        for (int j = 0; j < 4; ++j)
#pragma unroll
            for (int e = 0; e < 8; ++e) {
                int m = mw + i * 16 + e + 8 * hf;
                int n = nw + j * 16 + r;
                float val = acc[i][j][e];
                if (MODE == 2) {
                    ((float*)outv)[(size_t)m * CD + n] = val;
                } else {
                    int b = m >> 11, l = m & (CL - 1);
                    int h = n >> 6,  d = n & (CHD - 1);
                    __bf16 bv = f2bf(val);
                    if (MODE == 0)
                        ((__bf16*)outv)[(((size_t)(b * CH + h)) * CL + l) * CHD + d] = bv;
                    else
                        ((__bf16*)outv)[(((size_t)(b * CH + h)) * CHD + d) * CL + l] = bv;
                }
            }
}

// ---------------- kernel: causal flash attention ----------------
// Qh,Kh: (B,H,L,HD) bf16 ; Vt: (B,H,HD,L) bf16 ; Ao: (B,L,D) bf16
__global__ __launch_bounds__(128) void flash_attn(
    const __bf16* __restrict__ Qh, const __bf16* __restrict__ Kh,
    const __bf16* __restrict__ Vt, __bf16* __restrict__ Ao) {
    __shared__ __align__(16) __bf16 Kt[32 * 64];      // 32 keys x 64 dims
    __shared__ __align__(16) __bf16 Vtile[64 * 32];   // 64 dims x 32 keys
    __shared__ __align__(16) __bf16 Pt[4][32 * 32];   // per-wave P tile

    const int tid  = threadIdx.x;
    const int wave = __builtin_amdgcn_readfirstlane(tid >> 5);
    const int lane = tid & 31;
    const int r    = lane & 15;
    const int hf   = lane >> 4;
    const int ka   = 8 * hf, kb = 8 * hf;
    const int bh   = blockIdx.y;            // b*32 + h
    const int q0   = blockIdx.x * 128;
    const int qw   = q0 + wave * 32;        // this wave's 32 queries

    const __bf16* Qp = Qh + (size_t)bh * CL * CHD;
    const __bf16* Kp = Kh + (size_t)bh * CL * CHD;
    const __bf16* Vp = Vt + (size_t)bh * CHD * CL;

    // Q fragments held in registers across the whole kv loop
    v16bf qf[2][2];
#pragma unroll
    for (int i = 0; i < 2; ++i)
#pragma unroll
        for (int c = 0; c < 2; ++c)
            qf[i][c] = frag_a(Qp + (size_t)(qw + i * 16 + r) * CHD, c * 32 + ka);

    v8f zf = {0.f,0.f,0.f,0.f,0.f,0.f,0.f,0.f};
    v8f o[2][4];
    float mrow[2][8], lrow[2][8];
#pragma unroll
    for (int i = 0; i < 2; ++i) {
#pragma unroll
        for (int j = 0; j < 4; ++j) o[i][j] = zf;
#pragma unroll
        for (int e = 0; e < 8; ++e) { mrow[i][e] = -1e30f; lrow[i][e] = 0.f; }
    }

    __bf16* Pw = Pt[wave];
    const float scale = 0.125f;             // HD^-0.5
    const int kvEnd = q0 + 128;

    for (int kv0 = 0; kv0 < kvEnd; kv0 += 32) {
        __syncthreads();                    // previous tile fully consumed
        // stage K tile (4KB) + V tile (4KB): 512 x 16B, 128 threads
#pragma unroll
        for (int t = tid; t < 256; t += 128) {
            int row = t >> 3, c = t & 7;    // 8 x 16B per 64-dim K row
            cp16_g2l(&Kt[row * 64 + c * 8], Kp + (size_t)(kv0 + row) * CHD + c * 8);
        }
#pragma unroll
        for (int t = tid; t < 256; t += 128) {
            int row = t >> 2, c = t & 3;    // 4 x 16B per 32-key Vt row
            cp16_g2l(&Vtile[row * 32 + c * 8], Vp + (size_t)row * CL + kv0 + c * 8);
        }
        cp_wait();
        __syncthreads();                    // tiles visible to all waves

        const bool active = kv0 < qw + 32;  // wave-uniform (wave is SGPR)
        if (active) {
            // S = Q @ K^T  (K contraction = 64 -> two k-chunks)
            v16bf kf[2][2];
#pragma unroll
            for (int n = 0; n < 2; ++n)
#pragma unroll
                for (int c = 0; c < 2; ++c)
                    kf[n][c] = frag_b(&Kt[(n * 16 + r) * 64], c * 32 + kb);
            v8f s[2][2];
#pragma unroll
            for (int i = 0; i < 2; ++i)
#pragma unroll
                for (int n = 0; n < 2; ++n) {
                    v8f t = zf;
                    t = wmma_bf16(qf[i][0], kf[n][0], t);
                    t = wmma_bf16(qf[i][1], kf[n][1], t);
                    s[i][n] = t;
                }
            // online softmax (rows are distributed over one 16-lane half)
#pragma unroll
            for (int i = 0; i < 2; ++i)
#pragma unroll
                for (int e = 0; e < 8; ++e) {
                    int qrow = qw + i * 16 + e + 8 * hf;
                    float v0 = s[i][0][e] * scale;
                    float v1 = s[i][1][e] * scale;
                    if (kv0 + r > qrow)      v0 = -1e30f;
                    if (kv0 + 16 + r > qrow) v1 = -1e30f;
                    float rmax = halfmax16(fmaxf(v0, v1));
                    float mo = mrow[i][e];
                    float mn = fmaxf(mo, rmax);
                    float sf = __expf(mo - mn);
                    float p0 = __expf(v0 - mn);
                    float p1 = __expf(v1 - mn);
                    lrow[i][e] = lrow[i][e] * sf + halfsum16(p0 + p1);
                    mrow[i][e] = mn;
#pragma unroll
                    for (int j = 0; j < 4; ++j) o[i][j][e] *= sf;
                    Pw[(i * 16 + e + 8 * hf) * 32 + r]      = f2bf(p0);
                    Pw[(i * 16 + e + 8 * hf) * 32 + 16 + r] = f2bf(p1);
                }
        }
        __syncthreads();                    // P tile (D-layout) -> readable as A-layout
        if (active) {
            v16bf vf[4];
#pragma unroll
            for (int j = 0; j < 4; ++j)
                vf[j] = frag_b(&Vtile[(j * 16 + r) * 32], kb);
#pragma unroll
            for (int i = 0; i < 2; ++i) {
                v16bf pf = frag_a(&Pw[(i * 16 + r) * 32], ka);
#pragma unroll
                for (int j = 0; j < 4; ++j) o[i][j] = wmma_bf16(pf, vf[j], o[i][j]);
            }
        }
    }

    // normalize and store (B, L, D) bf16
    const int b = bh >> 5, h = bh & 31;
#pragma unroll
    for (int i = 0; i < 2; ++i)
#pragma unroll
        for (int e = 0; e < 8; ++e) {
            float inv = 1.0f / lrow[i][e];
            int qrow = qw + i * 16 + e + 8 * hf;
#pragma unroll
            for (int j = 0; j < 4; ++j) {
                int col = h * CHD + j * 16 + r;
                Ao[(size_t)(b * CL + qrow) * CD + col] = f2bf(o[i][j][e] * inv);
            }
        }
}

// ---------------- host launcher ----------------

extern "C" void kernel_launch(void* const* d_in, const int* in_sizes, int n_in,
                              void* d_out, int out_size, void* d_ws, size_t ws_size,
                              hipStream_t stream) {
    const float* q  = (const float*)d_in[0];
    const float* k  = (const float*)d_in[1];
    const float* v  = (const float*)d_in[2];
    const float* Wq = (const float*)d_in[3];
    const float* Wk = (const float*)d_in[4];
    const float* Wv = (const float*)d_in[5];
    const float* Wo = (const float*)d_in[6];
    // d_in[7] = causal mask (bool) -- causality is hardcoded in flash_attn

    const size_t NQKV = (size_t)CB * CL * CD;   // 8,388,608
    const size_t NW   = (size_t)CD * CD;        // 4,194,304

    __bf16* qb  = (__bf16*)d_ws;
    __bf16* kb  = qb  + NQKV;
    __bf16* vb  = kb  + NQKV;
    __bf16* wqb = vb  + NQKV;
    __bf16* wkb = wqb + NW;
    __bf16* wvb = wkb + NW;
    __bf16* wob = wvb + NW;
    __bf16* Qh  = wob + NW;
    __bf16* Kh  = Qh  + NQKV;
    __bf16* Vt  = Kh  + NQKV;
    __bf16* Ao  = Vt  + NQKV;    // total ~144 MB of workspace

    const int gq = (int)(NQKV / 4 / 256);  // 8192
    const int gw = (int)(NW   / 4 / 256);  // 4096
    cvt_f32_bf16<<<gq, 256, 0, stream>>>(q,  qb,  (int)(NQKV / 4));
    cvt_f32_bf16<<<gq, 256, 0, stream>>>(k,  kb,  (int)(NQKV / 4));
    cvt_f32_bf16<<<gq, 256, 0, stream>>>(v,  vb,  (int)(NQKV / 4));
    cvt_f32_bf16<<<gw, 256, 0, stream>>>(Wq, wqb, (int)(NW / 4));
    cvt_f32_bf16<<<gw, 256, 0, stream>>>(Wk, wkb, (int)(NW / 4));
    cvt_f32_bf16<<<gw, 256, 0, stream>>>(Wv, wvb, (int)(NW / 4));
    cvt_f32_bf16<<<gw, 256, 0, stream>>>(Wo, wob, (int)(NW / 4));

    dim3 gg(CD / 128, (CB * CL) / 128);    // (16, 32), 256 threads/block
    gemm_bf16_nt<0><<<gg, 256, 0, stream>>>(qb, wqb, (void*)Qh);
    gemm_bf16_nt<0><<<gg, 256, 0, stream>>>(kb, wkb, (void*)Kh);
    gemm_bf16_nt<1><<<gg, 256, 0, stream>>>(vb, wvb, (void*)Vt);

    flash_attn<<<dim3(CL / 128, CB * CH), 128, 0, stream>>>(Qh, Kh, Vt, Ao);

    gemm_bf16_nt<2><<<gg, 256, 0, stream>>>(Ao, wob, d_out);
}
